// ConditionalRandomField_52913997087452
// MI455X (gfx1250) — matrix operand order repaired and checked by reading, not compiled, not run on hardware
//
#include <hip/hip_runtime.h>
#include <hip/hip_bf16.h>
#include <stdint.h>

typedef __attribute__((ext_vector_type(16))) _Float16 v16h;
typedef __attribute__((ext_vector_type(8)))  _Float16 v8h;
typedef __attribute__((ext_vector_type(8)))  float    v8f;

constexpr int B = 128, L = 1024, T = 128;
constexpr int BROWS  = 16;       // batch rows per workgroup
constexpr int NT = T / 16;       // 8 N-tiles (one per wave)
constexpr int KC = T / 32;       // 4 K-chunks of 32

#define LOG2E 1.44269504088896340736f
#define LN2   0.69314718055994530942f
// raw v_exp_f32 / v_log_f32 (base-2); args here are bounded so no fixups needed
__device__ inline float fast_exp(float x) { return __builtin_amdgcn_exp2f(x * LOG2E); }
__device__ inline float fast_log(float x) { return __builtin_amdgcn_logf(x) * LN2; }

__device__ inline float max8(float4 a, float4 b) {
  return fmaxf(fmaxf(fmaxf(a.x, a.y), fmaxf(a.z, a.w)),
               fmaxf(fmaxf(b.x, b.y), fmaxf(b.z, b.w)));
}

// ---- WMMA fragment index maps (CDNA5 ISA 7.12.2, 16-bit A / B layouts) ----
// A (16x32 f16): lane<16: e0..7 -> K=e, e8..15 -> K=8+e ; lane>=16: +8
// K is contiguous over each run of 8 consecutive e values.
__device__ inline int a_k(int kc, int lane, int e) {
  int khi = (lane >= 16) ? 8 : 0;
  return kc * 32 + ((e < 8) ? (e + khi) : (8 + e + khi));
}
// B (32x16 f16): same K order as A: lane<16: e0..7 -> K=e, e8..15 -> K=8+e ; lane>=16: +8; N=lane&15
__device__ inline int b_k(int kc, int lane, int e) {
  return kc * 32 + ((lane >= 16) ? 8 : 0) + ((e < 8) ? e : (8 + e));
}

// ============================================================
// Forward algorithm (denominator): sequential scan over L with
// each step as an exp-domain matmul on v_wmma_f32_16x16x32_f16.
// One workgroup privately owns 16 batch rows for the whole scan.
// Two barriers per step; emit values AND mask bytes are register-
// pipelined one step ahead so no phase waits on a fresh VMEM load.
// ============================================================
__global__ __launch_bounds__(256)
void crf_forward_kernel(const float* __restrict__ inputs,
                        const unsigned char* __restrict__ mask,
                        const float* __restrict__ trans,
                        const float* __restrict__ start_t,
                        const float* __restrict__ end_t,
                        float* __restrict__ den /* [B] */) {
  __shared__ __align__(32) _Float16 expT_sw[KC * NT * 32 * 16]; // 32 KB, B-fragment order
  __shared__ __align__(32) _Float16 expA_sw[KC * 32 * 16];      //  4 KB, A-fragment order
  __shared__ __align__(16) float alpha[BROWS * T];              //  8 KB
  __shared__ __align__(16) float rowmax[BROWS];                 // full row maxima
  __shared__ __align__(16) float rowmax_part[BROWS][NT];        // per-wave column-max partials

  const int tid  = threadIdx.x;
  const int lane = tid & 31;
  const int wave = tid >> 5;
  const int b0   = blockIdx.x * BROWS;
  const int half = lane >> 4;     // 0/1
  const int l16  = lane & 15;

  // --- one-time: exp(transitions) pre-swizzled into WMMA B-fragment layout ---
  for (int g = tid; g < (KC * NT * 32 * 16) / 8; g += 256) {
    int ebase = (g & 1) * 8;
    int entry = g >> 1;                 // kc*NT*32 + nt*32 + ln
    int ln = entry & 31;
    int nt = (entry >> 5) & 7;
    int kc = entry >> 8;
    int K0 = b_k(kc, ln, ebase);        // contiguous run of 8
    int N  = nt * 16 + (ln & 15);
    v8h h;
    #pragma unroll
    for (int e = 0; e < 8; ++e)
      h[e] = (_Float16)fast_exp(trans[(K0 + e) * T + N]);
    *(v8h*)&expT_sw[g * 8] = h;
  }
  // --- alpha0 = start + logits[:,0,:] ---
  for (int idx = tid; idx < BROWS * T; idx += 256) {
    int r = idx >> 7;
    int c = idx & (T - 1);
    alpha[idx] = start_t[c] + inputs[(size_t)(b0 + r) * L * T + c];
  }
  __syncthreads();

  // --- seed rowmax_part from alpha0 (full scan once; butterfly -> all lanes) ---
  {
    int row = 2 * wave + half;
    const float4* ap = (const float4*)&alpha[row * T + l16 * 8];
    float m = max8(ap[0], ap[1]);
    #pragma unroll
    for (int st = 8; st >= 1; st >>= 1) m = fmaxf(m, __shfl_xor(m, st, 16));
    if (l16 < NT) rowmax_part[row][l16] = m;
  }
  __syncthreads();

  const v16h* Af = (const v16h*)expA_sw;
  const v16h* Bf = (const v16h*)expT_sw;

  // software-pipelined VMEM: emit (8 floats) + this WG's mask byte, one step ahead
  const float* ebase = inputs + (size_t)(b0 + half * 8) * L * T + wave * 16 + l16;
  const unsigned char* mbase = mask + (size_t)(b0 + l16) * L;   // per-lane row = l16
  float emit[8], emitN[8];
  unsigned char mkb, mkbN;
  #pragma unroll
  for (int r = 0; r < 8; ++r) emit[r] = ebase[(size_t)r * L * T + (size_t)1 * T];
  mkb = mbase[1];

  for (int t = 1; t < L; ++t) {
    // issue next step's loads as one clause; a full step of latency to hide
    {
      int tn = (t + 1 < L) ? t + 1 : t;
      #pragma unroll
      for (int r = 0; r < 8; ++r)
        emitN[r] = ebase[(size_t)r * L * T + (size_t)tn * T];
      mkbN = mbase[tn];
    }

    // (A) reduce rowmax partials privately; build shared A fragments (8 halves/thread)
    {
      int ebase8 = (tid & 1) * 8;
      int entry  = tid >> 1;            // kc*32 + ln
      int kc = entry >> 5;
      int ln = entry & 31;
      int r  = ln & 15;
      int K0 = a_k(kc, ln, ebase8);     // contiguous run of 8
      const float4* pp = (const float4*)&rowmax_part[r][0];
      float rm = max8(pp[0], pp[1]);
      const float4* ap = (const float4*)&alpha[r * T + K0];
      float4 x0 = ap[0], x1 = ap[1];
      v8h h;
      h[0] = (_Float16)fast_exp(x0.x - rm); h[1] = (_Float16)fast_exp(x0.y - rm);
      h[2] = (_Float16)fast_exp(x0.z - rm); h[3] = (_Float16)fast_exp(x0.w - rm);
      h[4] = (_Float16)fast_exp(x1.x - rm); h[5] = (_Float16)fast_exp(x1.y - rm);
      h[6] = (_Float16)fast_exp(x1.z - rm); h[7] = (_Float16)fast_exp(x1.w - rm);
      *(v8h*)&expA_sw[tid * 8] = h;
      if (tid < 16) {
        const float4* qq = (const float4*)&rowmax_part[tid][0];
        rowmax[tid] = max8(qq[0], qq[1]);
      }
    }
    __syncthreads();

    // (B) wave w: 16x16 N-tile, K=128 via 4x v_wmma_f32_16x16x32_f16
    v8f c = {};
    #pragma unroll
    for (int kc = 0; kc < KC; ++kc) {
      v16h a  = Af[kc * 32 + lane];
      v16h bb = Bf[(kc * NT + wave) * 32 + lane];
      c = __builtin_amdgcn_wmma_f32_16x16x32_f16(false, a, false, bb,
                                                 (short)0, c, false, false);
    }

    //     alpha' = rowmax + log(acc) + emit; per-wave ballot of pipelined mask
    //     bytes (wave-uniform), uniform fast path; next step's rowmax partials.
    {
      unsigned long long bal = __ballot((half == 0) && (mkb != 0));
      unsigned mbu = __builtin_amdgcn_readfirstlane((unsigned)bal);
      const float4* rp = (const float4*)&rowmax[half * 8];
      float4 r0 = rp[0], r1 = rp[1];
      float rmv[8] = {r0.x, r0.y, r0.z, r0.w, r1.x, r1.y, r1.z, r1.w};
      int col = wave * 16 + l16;
      float nv[8];
      #pragma unroll
      for (int r = 0; r < 8; ++r)
        nv[r] = rmv[r] + fast_log(c[r]) + emit[r];

      if (mbu == 0xFFFFu) {             // common case: one uniform branch, plain stores
        #pragma unroll
        for (int r = 0; r < 8; ++r)
          alpha[(r + half * 8) * T + col] = nv[r];
      } else {
        #pragma unroll
        for (int r = 0; r < 8; ++r)
          if (mbu & (1u << (r + half * 8))) alpha[(r + half * 8) * T + col] = nv[r];
      }

      #pragma unroll
      for (int r = 0; r < 8; ++r) {
        int row = r + half * 8;
        float pv = (mbu & (1u << row)) ? nv[r] : rmv[r];  // masked row unchanged -> old max
        #pragma unroll
        for (int st = 8; st >= 1; st >>= 1) pv = fmaxf(pv, __shfl_xor(pv, st, 16));
        if (l16 == 0) rowmax_part[row][wave] = pv;
      }
    }
    __syncthreads();

    #pragma unroll
    for (int r = 0; r < 8; ++r) emit[r] = emitN[r];
    mkb = mkbN;
  }

  // --- final: den[b] = logsumexp_j(alpha[b,j] + end_t[j]) ---
  {
    int row = 2 * wave + half;
    const float4* ap = (const float4*)&alpha[row * T + l16 * 8];
    float4 a0 = ap[0], a1 = ap[1];
    float v[8] = {a0.x, a0.y, a0.z, a0.w, a1.x, a1.y, a1.z, a1.w};
    float m = -3.4e38f;
    #pragma unroll
    for (int k = 0; k < 8; ++k) {
      v[k] += end_t[l16 * 8 + k];
      m = fmaxf(m, v[k]);
    }
    #pragma unroll
    for (int st = 8; st >= 1; st >>= 1) m = fmaxf(m, __shfl_xor(m, st, 16));
    float sum = 0.f;
    #pragma unroll
    for (int k = 0; k < 8; ++k) sum += fast_exp(v[k] - m);
    #pragma unroll
    for (int st = 8; st >= 1; st >>= 1) sum += __shfl_xor(sum, st, 16);
    if (l16 == 0) den[b0 + row] = m + fast_log(sum);
  }
}

// ============================================================
// Numerator (joint likelihood): one block per batch element.
// ============================================================
__global__ __launch_bounds__(256)
void crf_numerator_kernel(const float* __restrict__ inputs,
                          const long long* __restrict__ tags,
                          const unsigned char* __restrict__ mask,
                          const float* __restrict__ trans,
                          const float* __restrict__ start_t,
                          const float* __restrict__ end_t,
                          float* __restrict__ num /* [B] */) {
  __shared__ float ssum[256];
  __shared__ int   scnt[256];
  const int b = blockIdx.x, tid = threadIdx.x;
  const long long*     tg = tags + (size_t)b * L;
  const unsigned char* mk = mask + (size_t)b * L;
  const float*         lg = inputs + (size_t)b * L * T;

  float s = 0.f; int cnt = 0;
  for (int t = tid; t < L; t += 256) {
    bool m = (mk[t] != 0);
    cnt += m ? 1 : 0;
    if (t < L - 1) {
      float mf1 = (mk[t + 1] != 0) ? 1.f : 0.f;
      s += trans[(int)tg[t] * T + (int)tg[t + 1]] * mf1;       // transition scores
      s += lg[(size_t)t * T + (int)tg[t]] * (m ? 1.f : 0.f);   // emissions [0, L-2]
    }
  }
  ssum[tid] = s; scnt[tid] = cnt;
  __syncthreads();
  for (int st = 128; st >= 1; st >>= 1) {
    if (tid < st) { ssum[tid] += ssum[tid + st]; scnt[tid] += scnt[tid + st]; }
    __syncthreads();
  }
  if (tid == 0) {
    int last_idx = scnt[0] - 1;
    int last_tag = (int)tg[last_idx];
    float mfL = (mk[L - 1] != 0) ? 1.f : 0.f;
    num[b] = ssum[0] + start_t[(int)tg[0]] + end_t[last_tag]
           + lg[(size_t)(L - 1) * T + last_tag] * mfL;
  }
}

// Deterministic fixed-order final reduction: sum_b (num[b] - den[b])
__global__ void crf_final_reduce(const float* __restrict__ num,
                                 const float* __restrict__ den,
                                 float* __restrict__ out) {
  __shared__ float s[B];
  int tid = threadIdx.x;
  s[tid] = num[tid] - den[tid];
  __syncthreads();
  for (int st = B / 2; st >= 1; st >>= 1) {
    if (tid < st) s[tid] += s[tid + st];
    __syncthreads();
  }
  if (tid == 0) out[0] = s[0];
}

extern "C" void kernel_launch(void* const* d_in, const int* in_sizes, int n_in,
                              void* d_out, int out_size, void* d_ws, size_t ws_size,
                              hipStream_t stream) {
  const float*         inputs  = (const float*)d_in[0];
  const long long*     tags    = (const long long*)d_in[1];
  const unsigned char* mask    = (const unsigned char*)d_in[2];
  const float*         trans   = (const float*)d_in[3];
  const float*         start_t = (const float*)d_in[4];
  const float*         end_t   = (const float*)d_in[5];

  float* num = (float*)d_ws;     // [B]
  float* den = num + B;          // [B]

  crf_numerator_kernel<<<B, 256, 0, stream>>>(inputs, tags, mask, trans,
                                              start_t, end_t, num);
  crf_forward_kernel<<<B / BROWS, 256, 0, stream>>>(inputs, mask, trans,
                                                    start_t, end_t, den);
  crf_final_reduce<<<1, B, 0, stream>>>(num, den, (float*)d_out);
}
